// MultiHeadedAttention_63093069578848
// MI455X (gfx1250) — matrix operand order repaired and checked by reading, not compiled, not run on hardware
//
#include <hip/hip_runtime.h>
#include <hip/hip_bf16.h>

// ---------------------------------------------------------------------------
// MultiHeadedAttention forward for MI455X (gfx1250, wave32, WMMA).
// B=2, S=2048, D=1024, H=16, DH=64.
// Outputs: out (B,S,D) f32  followed by  attn (B,H,S,S) f32 in d_out.
// ---------------------------------------------------------------------------

typedef __bf16 v16bf __attribute__((ext_vector_type(16)));
typedef __bf16 v8bf  __attribute__((ext_vector_type(8)));
typedef float  v8f   __attribute__((ext_vector_type(8)));

#define NB 2
#define NS 2048
#define ND 1024
#define NH 16
#define NDH 64

__device__ __forceinline__ __bf16 f2bf(float f) {
  // round-to-nearest-even f32 -> bf16 via bit ops (no reliance on fp casts)
  unsigned u = __float_as_uint(f);
  unsigned r = (u + 0x7FFFu + ((u >> 16) & 1u)) >> 16;
  unsigned short h = (unsigned short)r;
  __bf16 b;
  __builtin_memcpy(&b, &h, 2);
  return b;
}

__device__ __forceinline__ v8f wmma_bf16(v16bf a, v16bf b, v8f c) {
  return __builtin_amdgcn_wmma_f32_16x16x32_bf16(false, a, false, b,
                                                 (short)0, c, false, false);
}

// ---------------------------------------------------------------------------
// f32 -> bf16 elementwise conversion (weights)
// ---------------------------------------------------------------------------
__global__ void cvt_bf16(const float* __restrict__ s, __bf16* __restrict__ d,
                         int n) {
  int i = blockIdx.x * 256 + threadIdx.x;
  if (i < n) d[i] = f2bf(s[i]);
}

// ---------------------------------------------------------------------------
// Projection GEMM:  Y[m, j] = sum_k A[m,k] * W[j,k] + bias[j]
// One wave per block; each wave produces a 16x64 output tile via
// v_wmma_f32_16x16x32_bf16, K-loop of 32.
// ABF:  A is bf16 (ctx) vs f32 (raw inputs, converted inline).
// MODE: 0 -> bf16 out, head-major [B,H,S,DH]   (Q, K)
//       1 -> bf16 out, head-transposed [B,H,DH,S]  (V^T)
//       2 -> f32 out, row-major [B*S, D]       (final output)
// ---------------------------------------------------------------------------
template <bool ABF, int MODE>
__global__ __launch_bounds__(32) void proj_gemm(const void* __restrict__ Ap,
                                                const __bf16* __restrict__ W,
                                                const float* __restrict__ bias,
                                                void* __restrict__ outp) {
  const int lane = threadIdx.x;
  const int r = lane & 15;    // column lane / row-of-A lane
  const int g = lane >> 4;    // half-wave group
  const int j0 = blockIdx.x * 64;
  const int m0 = blockIdx.y * 16;

  v8f acc[4] = {};

  for (int kk = 0; kk < ND; kk += 32) {
    // ---- A fragment: 16x32, row = r, K = g*16 + e
    v16bf a;
    if constexpr (ABF) {
      const __bf16* A = (const __bf16*)Ap;
      const __bf16* p0 = A + (size_t)(m0 + r) * ND + kk + g * 16;
      v8bf lo = *(const v8bf*)p0;
      v8bf hi = *(const v8bf*)(p0 + 8);
#pragma unroll
      for (int i = 0; i < 8; ++i) {
        a[i] = lo[i];
        a[i + 8] = hi[i];
      }
    } else {
      const float* A = (const float*)Ap;
      const float* p0 = A + (size_t)(m0 + r) * ND + kk + g * 16;
#pragma unroll
      for (int i = 0; i < 8; ++i) {
        a[i] = f2bf(p0[i]);
        a[i + 8] = f2bf(p0[i + 8]);
      }
    }
    // ---- B fragments: B[k,j] = W[j,k]; lane holds 16 contiguous k (32B load)
#pragma unroll
    for (int n = 0; n < 4; ++n) {
      const __bf16* bp = W + (size_t)(j0 + n * 16 + r) * ND + kk + g * 16;
      v16bf bfrag = *(const v16bf*)bp;
      acc[n] = wmma_bf16(a, bfrag, acc[n]);
    }
  }

#pragma unroll
  for (int n = 0; n < 4; ++n) {
    const int col = j0 + n * 16 + r;
    const float bj = bias[col];
#pragma unroll
    for (int v = 0; v < 8; ++v) {
      const int row = m0 + v + g * 8;  // global row in [0, B*S)
      const float y = acc[n][v] + bj;
      if constexpr (MODE == 2) {
        ((float*)outp)[(size_t)row * ND + col] = y;
      } else {
        const int b = row >> 11, s = row & (NS - 1);
        const int h = col >> 6, d = col & (NDH - 1);
        __bf16* o = (__bf16*)outp;
        if constexpr (MODE == 0)
          o[(((size_t)(b * NH + h)) * NS + s) * NDH + d] = f2bf(y);
        else
          o[(((size_t)(b * NH + h)) * NDH + d) * NS + s] = f2bf(y);
      }
    }
  }
}

// ---------------------------------------------------------------------------
// Fused attention: per (b, h, 16-row q tile), one wave.
// Pass 1: scores = (Q/sqrt(DH)) @ K^T via WMMA; mask; write raw scores into
//         attn; lane-local online softmax stats (each lane's C-fragment
//         column is a fixed key index).
// Merge:  shfl_xor tree -> per-lane (row = lane%16) max / inv-sum.
// Pass 2: reload scores in A-fragment layout straight from attn (L2-hot;
//         serves as the C->A layout transpose), normalize + zero pads,
//         rewrite attn, and accumulate ctx = P @ V with V stored transposed
//         so B-fragments are contiguous 32B loads.
// ---------------------------------------------------------------------------
__global__ __launch_bounds__(32) void attn_fused(
    const __bf16* __restrict__ qbf, const __bf16* __restrict__ kbf,
    const __bf16* __restrict__ vtb, const int* __restrict__ key_mask,
    float* __restrict__ attn, __bf16* __restrict__ ctxb) {
  const int lane = threadIdx.x;
  const int r = lane & 15;
  const int g = lane >> 4;
  const int qt = blockIdx.x, h = blockIdx.y, b = blockIdx.z;
  const int bh = b * NH + h;
  const int q0 = qt * 16;

  // Q A-fragments (16 x 64 -> two 16x32), loaded once, reused all kernel.
  const __bf16* qbase = qbf + ((size_t)bh * NS + q0) * NDH;
  v16bf aq[2];
#pragma unroll
  for (int s2 = 0; s2 < 2; ++s2) {
    const __bf16* p0 = qbase + (size_t)r * NDH + s2 * 32 + g * 16;
    v8bf lo = *(const v8bf*)p0;
    v8bf hi = *(const v8bf*)(p0 + 8);
#pragma unroll
    for (int i = 0; i < 8; ++i) {
      aq[s2][i] = lo[i];
      aq[s2][i + 8] = hi[i];
    }
  }

  const float scale = 0.125f;  // 1/sqrt(64)
  float m[8], sm[8];
#pragma unroll
  for (int v = 0; v < 8; ++v) {
    m[v] = -__builtin_inff();
    sm[v] = 0.f;
  }

  float* arow0 = attn + ((size_t)bh * NS + q0) * NS;

  // ---------------- pass 1: scores + online stats ----------------
  for (int kt = 0; kt < NS / 16; ++kt) {
    const int k0 = kt * 16;
    const __bf16* kr0 = kbf + ((size_t)bh * NS + k0 + r) * NDH + g * 16;
    if (kt + 1 < NS / 16) __builtin_prefetch(kr0 + 16 * NDH, 0, 3);
    v16bf b0 = *(const v16bf*)kr0;
    v16bf b1 = *(const v16bf*)(kr0 + 32);
    v8f acc = {};
    acc = wmma_bf16(aq[0], b0, acc);
    acc = wmma_bf16(aq[1], b1, acc);

    const int col = k0 + r;
    const bool mok = key_mask[(size_t)b * NS + col] != 0;
#pragma unroll
    for (int v = 0; v < 8; ++v) {
      const float s = mok ? acc[v] * scale : -1e18f;
      arow0[(size_t)(v + g * 8) * NS + col] = s;  // raw score (C layout)
      const float nm = fmaxf(m[v], s);
      sm[v] = sm[v] * __expf(m[v] - nm) + __expf(s - nm);
      m[v] = nm;
    }
  }

  // ---------------- merge stats across the 16-lane group ----------------
#pragma unroll
  for (int v = 0; v < 8; ++v) {
#pragma unroll
    for (int d = 1; d < 16; d <<= 1) {
      const float mo = __shfl_xor(m[v], d, 32);
      const float so = __shfl_xor(sm[v], d, 32);
      const float nm = fmaxf(m[v], mo);
      sm[v] = sm[v] * __expf(m[v] - nm) + so * __expf(mo - nm);
      m[v] = nm;
    }
  }
  // each lane extracts the stats for row (lane & 15)
  const int rg = (lane >> 3) & 1;  // group that owns my row
  float m_row = 0.f, s_row = 1.f;
#pragma unroll
  for (int v = 0; v < 8; ++v) {
    const float mo = __shfl_xor(m[v], 16, 32);
    const float so = __shfl_xor(sm[v], 16, 32);
    const float mm = (rg == g) ? m[v] : mo;
    const float ss = (rg == g) ? sm[v] : so;
    if ((lane & 7) == v) {
      m_row = mm;
      s_row = ss;
    }
  }
  const float inv_row = 1.0f / fmaxf(s_row, 1e-30f);

  // ---------------- pass 2: normalize + ctx = P @ V ----------------
  v8f cacc[4] = {};
  float* prow = attn + ((size_t)bh * NS + q0 + r) * NS;
  const __bf16* vbase = vtb + (size_t)bh * NDH * NS;

  for (int kp = 0; kp < NS / 32; ++kp) {
    const int kb0 = kp * 32;
    float* p0 = prow + kb0 + g * 16;  // K = g*16 + [0..8)
    float* p1 = p0 + 8;               // K = g*16 + 8 + [0..8)
    v16bf pa;
#pragma unroll
    for (int i = 0; i < 8; ++i) {
      const float s0 = p0[i], s1 = p1[i];
      const float e0 = (s0 <= -1e17f) ? 0.f : __expf(s0 - m_row) * inv_row;
      const float e1 = (s1 <= -1e17f) ? 0.f : __expf(s1 - m_row) * inv_row;
      p0[i] = e0;  // final attention probabilities (pads zeroed)
      p1[i] = e1;
      pa[i] = f2bf(e0);
      pa[i + 8] = f2bf(e1);
    }
    const __bf16* vb0 = vbase + kb0 + g * 16;
    if (kp + 1 < NS / 32) __builtin_prefetch(vb0 + 32, 0, 3);
#pragma unroll
    for (int n = 0; n < 4; ++n) {
      v16bf bv = *(const v16bf*)(vb0 + (size_t)(n * 16 + r) * NS);
      cacc[n] = wmma_bf16(pa, bv, cacc[n]);
    }
  }

  // write ctx (bf16, row-major [B*S, D]) for the output projection
#pragma unroll
  for (int n = 0; n < 4; ++n) {
#pragma unroll
    for (int v = 0; v < 8; ++v) {
      const int srow = q0 + v + g * 8;
      ctxb[((size_t)b * NS + srow) * ND + h * NDH + n * 16 + r] =
          f2bf(cacc[n][v]);
    }
  }
}

// ---------------------------------------------------------------------------
// host launch
// ---------------------------------------------------------------------------
extern "C" void kernel_launch(void* const* d_in, const int* in_sizes, int n_in,
                              void* d_out, int out_size, void* d_ws,
                              size_t ws_size, hipStream_t stream) {
  (void)in_sizes;
  (void)n_in;
  (void)out_size;
  (void)ws_size;

  const float* query = (const float*)d_in[0];
  const float* key = (const float*)d_in[1];
  const float* value = (const float*)d_in[2];
  const int* key_mask = (const int*)d_in[3];
  const float* Wq = (const float*)d_in[4];
  const float* bq = (const float*)d_in[5];
  const float* Wk = (const float*)d_in[6];
  const float* bk = (const float*)d_in[7];
  const float* Wv = (const float*)d_in[8];
  const float* bv = (const float*)d_in[9];
  const float* Wo = (const float*)d_in[10];
  const float* bo = (const float*)d_in[11];

  float* out = (float*)d_out;                       // (B,S,D)
  float* attn = out + (size_t)NB * NS * ND;         // (B,H,S,S)

  // workspace carve-up (bf16): 4 weights + Q + K + V^T + ctx  (~42 MB)
  __bf16* wqb = (__bf16*)d_ws;
  __bf16* wkb = wqb + (size_t)ND * ND;
  __bf16* wvb = wkb + (size_t)ND * ND;
  __bf16* wob = wvb + (size_t)ND * ND;
  __bf16* qbf = wob + (size_t)ND * ND;
  __bf16* kbf = qbf + (size_t)NB * NH * NS * NDH;
  __bf16* vtb = kbf + (size_t)NB * NH * NS * NDH;
  __bf16* ctx = vtb + (size_t)NB * NH * NS * NDH;

  const int nw = ND * ND;
  const int cvtBlocks = (nw + 255) / 256;
  cvt_bf16<<<cvtBlocks, 256, 0, stream>>>(Wq, wqb, nw);
  cvt_bf16<<<cvtBlocks, 256, 0, stream>>>(Wk, wkb, nw);
  cvt_bf16<<<cvtBlocks, 256, 0, stream>>>(Wv, wvb, nw);
  cvt_bf16<<<cvtBlocks, 256, 0, stream>>>(Wo, wob, nw);

  // projections: grid = (N/64, M/16) = (16, 256), 1 wave per block
  dim3 pgrid(ND / 64, (NB * NS) / 16);
  proj_gemm<false, 0><<<pgrid, 32, 0, stream>>>(query, wqb, bq, qbf);
  proj_gemm<false, 0><<<pgrid, 32, 0, stream>>>(key, wkb, bk, kbf);
  proj_gemm<false, 1><<<pgrid, 32, 0, stream>>>(value, wvb, bv, vtb);

  // fused attention: grid = (S/16, H, B), 1 wave per block
  dim3 agrid(NS / 16, NH, NB);
  attn_fused<<<agrid, 32, 0, stream>>>(qbf, kbf, vtb, key_mask, attn, ctx);

  // output projection: ctx(bf16) @ Wo^T + bo -> f32 out
  proj_gemm<true, 2><<<pgrid, 32, 0, stream>>>(ctx, wob, bo, out);
}
